// ConvCapsule_77541339562046
// MI455X (gfx1250) — matrix operand order repaired and checked by reading, not compiled, not run on hardware
//
#include <hip/hip_runtime.h>

// ConvCapsule + dynamic routing, fully fused per output position (MI455X / gfx1250).
//
// Kernel 1 (prep): pack conv weights (512x72 f32) into WMMA-B-fragment-ordered,
//   K-padded (72->96) f16 buffer in d_ws: Wpack[nt][kt][lane][16 halves].
//   96 KB total -> L2 resident, shared by all 3136 blocks.
// Kernel 2 (fused): one workgroup per (b,ho,wo). 256 threads = 8 wave32.
//   Phase 1: u_hat(32x512) = A(32x72 im2col) x W^T via v_wmma_f32_16x16x32_f16,
//            accumulators f32, result kept in 64 KB LDS tile.
//   Phase 2: 3 dynamic-routing iterations entirely in LDS; write squash(s).

#define BATCH    16
#define HW       14
#define IN_CAPS  32
#define IN_DIM   8
#define OUT_CAPS 32
#define OUT_DIM  16
#define NCOLS    (OUT_CAPS * OUT_DIM)   // 512
#define KDIM     (IN_DIM * 9)           // 72
#define KPAD     96                     // 3 WMMA k-steps of 32
#define NT_TILES 32                     // 512 / 16
#define KT_STEPS 3
#define RITERS   3

typedef __attribute__((ext_vector_type(16))) _Float16 v16h;
typedef __attribute__((ext_vector_type(8)))  float    v8f;

union AFrag { v16h v; _Float16 h[16]; };

// ---------------------------------------------------------------------------
// Prep: Wpack[((nt*3 + kt)*32 + lane)*16 + i]
//   lane<16 : col N = nt*16+lane,    K = kt*32 + i        (i = 0..15)
//   lane>=16: col N = nt*16+lane-16, K = kt*32 + 16 + i
//   K >= 72 -> 0 (pad). One 32-byte chunk per thread, coalesced.
// ---------------------------------------------------------------------------
__global__ __launch_bounds__(256) void pack_weights(
    const float* __restrict__ Wg, _Float16* __restrict__ Wp)
{
  const int t = blockIdx.x * 256 + threadIdx.x;       // 0 .. 3071
  if (t >= NT_TILES * KT_STEPS * 32) return;
  const int lane = t & 31;
  const int kt   = (t >> 5) % KT_STEPS;
  const int nt   = t / (KT_STEPS * 32);
  const int col   = nt * 16 + (lane & 15);
  const int kbase = kt * 32 + ((lane & 16) ? 16 : 0);
  AFrag f;
  #pragma unroll
  for (int i = 0; i < 16; ++i) {
    const int k = kbase + i;
    f.h[i] = (k < KDIM) ? (_Float16)Wg[col * KDIM + k] : (_Float16)0.f;
  }
  *((v16h*)(Wp + (size_t)t * 16)) = f.v;
}

// ---------------------------------------------------------------------------
// Fused conv-capsule + routing
// ---------------------------------------------------------------------------
__global__ __launch_bounds__(256) void convcaps_fused(
    const float* __restrict__ x,        // (B, H, W, IC, D)
    const _Float16* __restrict__ Wp,    // packed fragments (see above)
    float* __restrict__ out)            // (B, Ho, Wo, OC, OD)
{
  __shared__ __attribute__((aligned(16))) _Float16 Alds[IN_CAPS * KPAD]; // 6 KB
  __shared__ __attribute__((aligned(16))) float uhat[IN_CAPS * NCOLS];   // 64 KB
  __shared__ float blog[IN_CAPS * OUT_CAPS];                             // 4 KB
  __shared__ float cpro[IN_CAPS * OUT_CAPS];                             // 4 KB
  __shared__ float sv[NCOLS];                                            // 2 KB
  __shared__ float scl[OUT_CAPS];

  const int tid  = threadIdx.x;
  const int lane = tid & 31;
  const int wave = tid >> 5;
  const int pos  = blockIdx.x;
  const int wo   = pos % HW;
  const int ho   = (pos / HW) % HW;
  const int b    = pos / (HW * HW);

  // ---------------- Phase 0: im2col patch -> Alds (f16), K padded ---------
  for (int i = tid; i < IN_CAPS * (KPAD - KDIM); i += 256) {
    int r = i / (KPAD - KDIM), c = KDIM + i % (KPAD - KDIM);
    Alds[r * KPAD + c] = (_Float16)0.f;
  }
  for (int task = tid; task < IN_CAPS * 9; task += 256) {
    int ic = task / 9, kk = task % 9;
    int kh = kk / 3, kw = kk % 3;
    int h = ho - 1 + kh, w = wo - 1 + kw;
    bool inb = (h >= 0) && (h < HW) && (w >= 0) && (w < HW);
    const float* src = x + (((size_t)(b * HW + h) * HW + w) * IN_CAPS + ic) * IN_DIM;
    #pragma unroll
    for (int d = 0; d < IN_DIM; ++d) {
      float val = inb ? src[d] : 0.f;
      Alds[ic * KPAD + d * 9 + kk] = (_Float16)val;   // k = d*9 + kh*3 + kw
    }
  }
  for (int i = tid; i < IN_CAPS * OUT_CAPS; i += 256) blog[i] = 0.f;
  __syncthreads();

  // ---------------- Phase 1: WMMA GEMM -> uhat (LDS) ----------------------
  // Load all 6 A fragments (2 M-tiles x 3 k-steps) once per wave: 12 ds_load_b128.
  AFrag a[2][KT_STEPS];
  #pragma unroll
  for (int mt = 0; mt < 2; ++mt) {
    #pragma unroll
    for (int kt = 0; kt < KT_STEPS; ++kt) {
      const int row  = mt * 16 + (lane & 15);
      const int base = row * KPAD + kt * 32 + ((lane & 16) ? 16 : 0);
      #pragma unroll
      for (int i = 0; i < 8; ++i) a[mt][kt].h[i]     = Alds[base + i];
      #pragma unroll
      for (int i = 0; i < 8; ++i) a[mt][kt].h[8 + i] = Alds[base + 8 + i];
    }
  }

  const v16h* __restrict__ Wfrag = (const v16h*)Wp;   // 32B fragments
  #pragma unroll
  for (int j = 0; j < 4; ++j) {
    const int nt = wave + j * 8;                      // 8 waves cover 32 N-tiles
    // 3 B fragments for this N-tile: aligned 32B loads, reused by both M-tiles
    v16h b0 = Wfrag[(nt * KT_STEPS + 0) * 32 + lane];
    v16h b1 = Wfrag[(nt * KT_STEPS + 1) * 32 + lane];
    v16h b2 = Wfrag[(nt * KT_STEPS + 2) * 32 + lane];
    #pragma unroll
    for (int mt = 0; mt < 2; ++mt) {
      v8f acc = {};
      acc = __builtin_amdgcn_wmma_f32_16x16x32_f16(false, a[mt][0].v, false, b0,
                                                   (short)0, acc, false, false);
      acc = __builtin_amdgcn_wmma_f32_16x16x32_f16(false, a[mt][1].v, false, b1,
                                                   (short)0, acc, false, false);
      acc = __builtin_amdgcn_wmma_f32_16x16x32_f16(false, a[mt][2].v, false, b2,
                                                   (short)0, acc, false, false);
      // C layout: VGPR r -> (M = r (+8 for hi lanes), N = lane&15)
      const int col   = nt * 16 + (lane & 15);
      const int rbase = mt * 16 + ((lane & 16) ? 8 : 0);
      #pragma unroll
      for (int r = 0; r < 8; ++r)
        uhat[(rbase + r) * NCOLS + col] = acc[r];
    }
  }
  __syncthreads();

  // ---------------- Phase 2: dynamic routing in LDS -----------------------
  for (int it = 0; it < RITERS; ++it) {
    // c[ic][:] = softmax over output capsules of b[ic][:]
    if (tid < IN_CAPS) {
      float m = -1e30f;
      #pragma unroll
      for (int oc = 0; oc < OUT_CAPS; ++oc)
        m = fmaxf(m, blog[tid * OUT_CAPS + oc]);
      float e[OUT_CAPS];
      float sum = 0.f;
      #pragma unroll
      for (int oc = 0; oc < OUT_CAPS; ++oc) {
        e[oc] = __expf(blog[tid * OUT_CAPS + oc] - m);
        sum += e[oc];
      }
      float inv = 1.f / sum;
      #pragma unroll
      for (int oc = 0; oc < OUT_CAPS; ++oc)
        cpro[tid * OUT_CAPS + oc] = e[oc] * inv;
    }
    __syncthreads();

    // s[n] = sum_ic uhat[ic][n] * c[ic][n/16]
    #pragma unroll
    for (int rep = 0; rep < 2; ++rep) {
      const int n  = tid + rep * 256;
      const int oc = n >> 4;
      float s = 0.f;
      #pragma unroll 8
      for (int ic = 0; ic < IN_CAPS; ++ic)
        s += uhat[ic * NCOLS + n] * cpro[ic * OUT_CAPS + oc];
      sv[n] = s;
    }
    __syncthreads();

    // squash scale per output capsule: sq/(1+sq)/(sqrt(sq)+1e-6)
    if (tid < OUT_CAPS) {
      float sq = 0.f;
      #pragma unroll
      for (int od = 0; od < OUT_DIM; ++od) {
        float t = sv[tid * OUT_DIM + od];
        sq += t * t;
      }
      scl[tid] = sq / (1.f + sq) / (sqrtf(sq) + 1e-6f);
    }
    __syncthreads();

    if (it == RITERS - 1) break;

    // b[ic][oc] += sum_od uhat[ic][oc*16+od] * (scale[oc] * s[oc*16+od])
    for (int p = tid; p < IN_CAPS * OUT_CAPS; p += 256) {
      const int ic = p >> 5, oc = p & 31;
      const float sc = scl[oc];
      float dot = 0.f;
      #pragma unroll
      for (int od = 0; od < OUT_DIM; ++od)
        dot += uhat[ic * NCOLS + oc * OUT_DIM + od] * (sc * sv[oc * OUT_DIM + od]);
      blog[p] += dot;
    }
    __syncthreads();
  }

  // final v = scale * s -> global out (B, Ho, Wo, OC, OD)
  #pragma unroll
  for (int rep = 0; rep < 2; ++rep) {
    const int n = tid + rep * 256;
    out[(size_t)pos * NCOLS + n] = sv[n] * scl[n >> 4];
  }
}

extern "C" void kernel_launch(void* const* d_in, const int* in_sizes, int n_in,
                              void* d_out, int out_size, void* d_ws, size_t ws_size,
                              hipStream_t stream) {
  const float* x  = (const float*)d_in[0];
  const float* Wg = (const float*)d_in[1];
  float* out = (float*)d_out;
  _Float16* Wp = (_Float16*)d_ws;          // 32*3*32*16 halves = 96 KB

  pack_weights<<<dim3(12), dim3(256), 0, stream>>>(Wg, Wp);
  convcaps_fused<<<dim3(BATCH * HW * HW), dim3(256), 0, stream>>>(x, Wp, out);
}
